// MyMultiAttention_18391049961929
// MI455X (gfx1250) — matrix operand order repaired and checked by reading, compiled, not run on hardware
//
#include <hip/hip_runtime.h>

// Problem constants (from reference): B=8, S=1024, D=768, H=12, E=64
#define BB 8
#define SS 1024
#define DD 768
#define HH 12
#define EE 64

typedef _Float16 half_t;
typedef __attribute__((ext_vector_type(16))) _Float16 v16h;
typedef __attribute__((ext_vector_type(8)))  float    v8f;

// ---------------------------------------------------------------------------
// WMMA fragment loaders (wave32, contiguous k=16*(lane/16)+s for A and B, CDNA5 ISA §7.12.2)
// A (16x32 f16): lanes 0-15 -> M=lane, K=0..15; lanes 16-31 ->
//   M=lane-16, K=16..31; VGPR j holds a packed K pair.
// B (32x16 f16) stored transposed Bt[n][k]: n=lane%16; lanes 0-15 K=0..15,
//   lanes 16-31 K=16..31; VGPR j holds K=2j,2j+1 (contiguous b32 pair).
// C/D (16x16 f32): VGPR r: lanes 0-15 -> (M=r,N=lane); lanes 16-31 -> M=r+8.
// ---------------------------------------------------------------------------
__device__ __forceinline__ v16h load_a_frag(const half_t* __restrict__ base,
                                            int ld, int lane) {
  int row   = lane & 15;
  int khalf = (lane >> 4) << 4;
  const half_t* p = base + (size_t)row * ld;
  v16h a;
#pragma unroll
  for (int j = 0; j < 8; ++j) {
    int kk = 2 * j + khalf;
    a[2 * j]     = p[kk];
    a[2 * j + 1] = p[kk + 1];
  }
  return a;
}

__device__ __forceinline__ v16h load_bt_frag(const half_t* __restrict__ base,
                                             int ld, int lane) {
  int n  = lane & 15;
  int kb = (lane >> 4) << 4;
  const half_t* p = base + (size_t)n * ld + kb;
  v16h b;
#pragma unroll
  for (int j = 0; j < 8; ++j) {
    b[2 * j]     = p[2 * j];
    b[2 * j + 1] = p[2 * j + 1];
  }
  return b;
}

#define WMMA_F16(a, b, c) \
  __builtin_amdgcn_wmma_f32_16x16x32_f16(false, (a), false, (b), (short)0, (c), false, false)

// ---------------------------------------------------------------------------
// CDNA5 async memory->LDS DMA (CDNA5 ISA op 98, ASYNCcnt).
// The LDS operand is the low 32 bits of the flat pointer (flat->LDS lowering
// truncates addr[31:0]); global operand is the 64-bit VA. 16B alignment req'd.
// ---------------------------------------------------------------------------
__device__ __forceinline__ void async_copy_b128(const void* g, void* l) {
  unsigned lds_off = (unsigned)(unsigned long long)l;
  asm volatile("global_load_async_to_lds_b128 %0, %1, off"
               :: "v"(lds_off), "v"((unsigned long long)g)
               : "memory");
}

__device__ __forceinline__ void wait_asynccnt0() {
  asm volatile("s_wait_asynccnt 0x0" ::: "memory");
}

// ---------------------------------------------------------------------------
// Conversion kernels: fp32 -> f16, with transposes so GEMMs see Bt layout.
// ---------------------------------------------------------------------------
__global__ void cvt_x_kernel(const float* __restrict__ x, half_t* __restrict__ xh, int n) {
  int i = blockIdx.x * blockDim.x + threadIdx.x;
  if (i < n) xh[i] = (half_t)x[i];
}

// W[h][c][d][e] -> wt[hc][e][d]
__global__ void cvt_w_kernel(const float* __restrict__ W, half_t* __restrict__ wt) {
  int i = blockIdx.x * blockDim.x + threadIdx.x;
  const int total = HH * 3 * EE * DD;
  if (i >= total) return;
  int d  = i % DD;
  int t  = i / DD;
  int e  = t % EE;
  int hc = t / EE;
  wt[i] = (half_t)W[((size_t)hc * DD + d) * EE + e];
}

// Wo[j][e] -> woT[e][j]
__global__ void cvt_wo_kernel(const float* __restrict__ Wo, half_t* __restrict__ woT) {
  int i = blockIdx.x * blockDim.x + threadIdx.x;
  const int total = EE * HH * EE;
  if (i >= total) return;
  int j = i % (HH * EE);
  int e = i / (HH * EE);
  woT[i] = (half_t)Wo[(size_t)j * EE + e];
}

// ---------------------------------------------------------------------------
// QKV GEMM, software-pipelined: distinct fragment registers per stage so the
// scheduler can clause the loads and overlap them with the 4 WMMAs.
// ---------------------------------------------------------------------------
__global__ void __launch_bounds__(128)
qkv_kernel(const half_t* __restrict__ xh, const half_t* __restrict__ wt,
           half_t* __restrict__ q, half_t* __restrict__ k, half_t* __restrict__ vT) {
  int b    = blockIdx.x;   // 0..7
  int hc   = blockIdx.y;   // 0..35
  int mblk = blockIdx.z;   // 0..15
  int h = hc / 3, c = hc % 3;
  int wave = threadIdx.x >> 5, lane = threadIdx.x & 31;
  int m0 = mblk * 64 + wave * 16;

  const half_t* A  = xh + ((size_t)b * SS + m0) * DD;
  const half_t* Bt = wt + (size_t)hc * EE * DD;

  v8f acc[4] = {};
  v16h a  = load_a_frag(A, DD, lane);
  v16h b0 = load_bt_frag(Bt + (size_t)0  * DD, DD, lane);
  v16h b1 = load_bt_frag(Bt + (size_t)16 * DD, DD, lane);
  v16h b2 = load_bt_frag(Bt + (size_t)32 * DD, DD, lane);
  v16h b3 = load_bt_frag(Bt + (size_t)48 * DD, DD, lane);

  for (int kk = 32; kk < DD; kk += 32) {
    v16h an  = load_a_frag(A + kk, DD, lane);
    v16h bn0 = load_bt_frag(Bt + (size_t)0  * DD + kk, DD, lane);
    v16h bn1 = load_bt_frag(Bt + (size_t)16 * DD + kk, DD, lane);
    v16h bn2 = load_bt_frag(Bt + (size_t)32 * DD + kk, DD, lane);
    v16h bn3 = load_bt_frag(Bt + (size_t)48 * DD + kk, DD, lane);
    __builtin_prefetch(A + kk + 64, 0, 1);   // -> global_prefetch_b8
    acc[0] = WMMA_F16(a, b0, acc[0]);
    acc[1] = WMMA_F16(a, b1, acc[1]);
    acc[2] = WMMA_F16(a, b2, acc[2]);
    acc[3] = WMMA_F16(a, b3, acc[3]);
    a = an; b0 = bn0; b1 = bn1; b2 = bn2; b3 = bn3;
  }
  acc[0] = WMMA_F16(a, b0, acc[0]);
  acc[1] = WMMA_F16(a, b1, acc[1]);
  acc[2] = WMMA_F16(a, b2, acc[2]);
  acc[3] = WMMA_F16(a, b3, acc[3]);

  int half_sel = lane >> 4;
  int ncol     = lane & 15;
  size_t bh = (size_t)b * HH + h;
#pragma unroll
  for (int j = 0; j < 4; ++j) {
#pragma unroll
    for (int r = 0; r < 8; ++r) {
      int row = m0 + r + half_sel * 8;
      int col = j * 16 + ncol;
      half_t val = (half_t)acc[j][r];
      if (c == 0)      q [(bh * SS + row) * EE + col] = val;
      else if (c == 1) k [(bh * SS + row) * EE + col] = val;
      else             vT[(bh * EE + col) * SS + row] = val;  // v stored transposed
    }
  }
}

// ---------------------------------------------------------------------------
// Flash attention: 4 waves, 64 query rows per block, KV tiles of 64 staged in
// ping-pong LDS buffers via GLOBAL_LOAD_ASYNC_TO_LDS_B128 (ASYNCcnt), one
// barrier per tile. Online softmax; reference scale = /(E*0.5) = /32.
// ---------------------------------------------------------------------------
__global__ void __launch_bounds__(128)
attn_kernel(const half_t* __restrict__ q, const half_t* __restrict__ k,
            const half_t* __restrict__ vT, half_t* __restrict__ oc) {
  __shared__ half_t kt[2][64][72];      // k tile  [t][e]  (= Bt layout for q@k^T)
  __shared__ half_t vt[2][64][72];      // vT tile [e][t]  (= Bt layout for p@v)
  __shared__ half_t pbuf[4][16][72];    // per-wave P staging (C-layout -> A-layout)

  int bh   = blockIdx.x;   // 0..95
  int mblk = blockIdx.y;   // 0..15
  int b = bh / HH, h = bh % HH;
  int wave = threadIdx.x >> 5, lane = threadIdx.x & 31;
  int tid  = threadIdx.x;
  int m0 = mblk * 64 + wave * 16;

  const half_t* qbase = q  + (size_t)bh * SS * EE;
  const half_t* kbase = k  + (size_t)bh * SS * EE;
  const half_t* vbase = vT + (size_t)bh * EE * SS;

  // Each thread DMAs 64B of k-tile and 64B of v-tile (4+4 b128, all 16B-aligned).
  int crow = tid >> 1;             // 0..63
  int ccol = (tid & 1) * 32;       // 0 or 32 (halves)
  auto issue_tile = [&](int buf, int t0) {
    const half_t* kg = kbase + ((size_t)(t0 + crow)) * EE + ccol;
    const half_t* vg = vbase + (size_t)crow * SS + t0 + ccol;
#pragma unroll
    for (int cc = 0; cc < 32; cc += 8) {
      async_copy_b128(kg + cc, &kt[buf][crow][ccol + cc]);
      async_copy_b128(vg + cc, &vt[buf][crow][ccol + cc]);
    }
  };

  // Preload this wave's 16 query rows as two A-fragments (e in [0,32),[32,64)).
  v16h qa0 = load_a_frag(qbase + (size_t)m0 * EE,      EE, lane);
  v16h qa1 = load_a_frag(qbase + (size_t)m0 * EE + 32, EE, lane);

  v8f acc[4] = {};
  float mrow[8], lrow[8];
#pragma unroll
  for (int r = 0; r < 8; ++r) { mrow[r] = -1e30f; lrow[r] = 0.0f; }

  const float scale = 1.0f / (EE * 0.5f);

  issue_tile(0, 0);

  for (int it = 0; it < SS / 64; ++it) {
    int cur = it & 1;
    wait_asynccnt0();      // our async DMAs for tile `it` have landed in LDS
    __syncthreads();       // everyone's have
    if (it + 1 < SS / 64) issue_tile(cur ^ 1, (it + 1) * 64);  // overlap next DMA

    // S = q @ k^T  (4 n-tiles over t; K=64 over e in 2 WMMAs each)
    v8f sacc[4] = {};
#pragma unroll
    for (int j = 0; j < 4; ++j) {
      v16h kb0 = load_bt_frag(&kt[cur][j * 16][0],  72, lane);
      v16h kb1 = load_bt_frag(&kt[cur][j * 16][32], 72, lane);
      sacc[j] = WMMA_F16(qa0, kb0, sacc[j]);
      sacc[j] = WMMA_F16(qa1, kb1, sacc[j]);
    }

    // Online softmax. Row r's 64 scores live in one 16-lane half across the 4
    // n-tiles; xor-butterfly d in {1,2,4,8} reduces exactly within the half.
#pragma unroll
    for (int r = 0; r < 8; ++r) {
      float mx = -1e30f;
#pragma unroll
      for (int j = 0; j < 4; ++j) {
        float s = sacc[j][r] * scale;
        sacc[j][r] = s;
        mx = fmaxf(mx, s);
      }
#pragma unroll
      for (int d = 1; d < 16; d <<= 1) mx = fmaxf(mx, __shfl_xor(mx, d, 32));
      float mnew  = fmaxf(mrow[r], mx);
      float alpha = __expf(mrow[r] - mnew);
      mrow[r] = mnew;
      float rs = 0.0f;
#pragma unroll
      for (int j = 0; j < 4; ++j) {
        float p = __expf(sacc[j][r] - mnew);
        sacc[j][r] = p;
        rs += p;
      }
#pragma unroll
      for (int d = 1; d < 16; d <<= 1) rs += __shfl_xor(rs, d, 32);
      lrow[r] = lrow[r] * alpha + rs;
#pragma unroll
      for (int j = 0; j < 4; ++j) acc[j][r] *= alpha;
    }

    // C-layout -> row-major P in wave-private LDS (same-wave LDS is in-order).
    {
      int rowadd = (lane >> 4) * 8;
      int ncol   = lane & 15;
#pragma unroll
      for (int r = 0; r < 8; ++r)
#pragma unroll
        for (int j = 0; j < 4; ++j)
          pbuf[wave][r + rowadd][j * 16 + ncol] = (half_t)sacc[j][r];
    }

    // O += P @ V
    v16h pa0 = load_a_frag(&pbuf[wave][0][0],  72, lane);
    v16h pa1 = load_a_frag(&pbuf[wave][0][32], 72, lane);
#pragma unroll
    for (int j = 0; j < 4; ++j) {
      v16h bv0 = load_bt_frag(&vt[cur][j * 16][0],  72, lane);
      v16h bv1 = load_bt_frag(&vt[cur][j * 16][32], 72, lane);
      acc[j] = WMMA_F16(pa0, bv0, acc[j]);
      acc[j] = WMMA_F16(pa1, bv1, acc[j]);
    }
  }

  // Normalize and write concatenated heads: oc[b][s][h*E + e].
  int half_sel = lane >> 4;
  int ncol     = lane & 15;
#pragma unroll
  for (int r = 0; r < 8; ++r) {
    float inv = 1.0f / lrow[r];
    int row = m0 + r + half_sel * 8;
#pragma unroll
    for (int j = 0; j < 4; ++j)
      oc[((size_t)b * SS + row) * (HH * EE) + h * EE + j * 16 + ncol] =
          (half_t)(acc[j][r] * inv);
  }
}

// ---------------------------------------------------------------------------
// Output projection: [B*S x 768] @ Wo[768 x 64] -> f32 out. Same pipelining.
// ---------------------------------------------------------------------------
__global__ void __launch_bounds__(128)
proj_kernel(const half_t* __restrict__ oc, const half_t* __restrict__ woT,
            float* __restrict__ out) {
  int mblk = blockIdx.x;   // 0..127
  int wave = threadIdx.x >> 5, lane = threadIdx.x & 31;
  int m0 = mblk * 64 + wave * 16;
  const int K = HH * EE;   // 768

  const half_t* A = oc + (size_t)m0 * K;
  v8f acc[4] = {};
  v16h a  = load_a_frag(A, K, lane);
  v16h b0 = load_bt_frag(woT + (size_t)0  * K, K, lane);
  v16h b1 = load_bt_frag(woT + (size_t)16 * K, K, lane);
  v16h b2 = load_bt_frag(woT + (size_t)32 * K, K, lane);
  v16h b3 = load_bt_frag(woT + (size_t)48 * K, K, lane);

  for (int kk = 32; kk < K; kk += 32) {
    v16h an  = load_a_frag(A + kk, K, lane);
    v16h bn0 = load_bt_frag(woT + (size_t)0  * K + kk, K, lane);
    v16h bn1 = load_bt_frag(woT + (size_t)16 * K + kk, K, lane);
    v16h bn2 = load_bt_frag(woT + (size_t)32 * K + kk, K, lane);
    v16h bn3 = load_bt_frag(woT + (size_t)48 * K + kk, K, lane);
    acc[0] = WMMA_F16(a, b0, acc[0]);
    acc[1] = WMMA_F16(a, b1, acc[1]);
    acc[2] = WMMA_F16(a, b2, acc[2]);
    acc[3] = WMMA_F16(a, b3, acc[3]);
    a = an; b0 = bn0; b1 = bn1; b2 = bn2; b3 = bn3;
  }
  acc[0] = WMMA_F16(a, b0, acc[0]);
  acc[1] = WMMA_F16(a, b1, acc[1]);
  acc[2] = WMMA_F16(a, b2, acc[2]);
  acc[3] = WMMA_F16(a, b3, acc[3]);

  int half_sel = lane >> 4;
  int ncol     = lane & 15;
#pragma unroll
  for (int j = 0; j < 4; ++j)
#pragma unroll
    for (int r = 0; r < 8; ++r)
      out[(size_t)(m0 + r + half_sel * 8) * EE + j * 16 + ncol] = acc[j][r];
}

// ---------------------------------------------------------------------------
extern "C" void kernel_launch(void* const* d_in, const int* in_sizes, int n_in,
                              void* d_out, int out_size, void* d_ws, size_t ws_size,
                              hipStream_t stream) {
  (void)in_sizes; (void)n_in; (void)out_size; (void)ws_size;
  const float* x  = (const float*)d_in[0];   // [B,S,D]
  const float* W  = (const float*)d_in[1];   // [H,3,D,E]
  const float* Wo = (const float*)d_in[2];   // [H*E,E]
  float* out = (float*)d_out;                // [B,S,E]

  char* ws = (char*)d_ws;
  size_t off = 0;
  auto take = [&](size_t n_half) {
    half_t* p = (half_t*)(ws + off);
    off += ((n_half * sizeof(half_t) + 255) / 256) * 256;
    return p;
  };
  half_t* xh  = take((size_t)BB * SS * DD);        // f16 x
  half_t* wt  = take((size_t)HH * 3 * EE * DD);    // f16 W^T per (h,c)
  half_t* woT = take((size_t)EE * HH * EE);        // f16 Wo^T
  half_t* qd  = take((size_t)BB * HH * SS * EE);   // q [b,h,s,e]
  half_t* kd  = take((size_t)BB * HH * SS * EE);   // k [b,h,s,e]
  half_t* vT  = take((size_t)BB * HH * EE * SS);   // v^T [b,h,e,s]
  half_t* oc  = take((size_t)BB * SS * HH * EE);   // concat heads [b,s,h*e]

  {
    int n = BB * SS * DD;
    cvt_x_kernel<<<(n + 255) / 256, 256, 0, stream>>>(x, xh, n);
  }
  {
    int n = HH * 3 * EE * DD;
    cvt_w_kernel<<<(n + 255) / 256, 256, 0, stream>>>(W, wt);
  }
  {
    int n = EE * HH * EE;
    cvt_wo_kernel<<<(n + 255) / 256, 256, 0, stream>>>(Wo, woT);
  }

  qkv_kernel <<<dim3(BB, HH * 3, SS / 64), 128, 0, stream>>>(xh, wt, qd, kd, vT);
  attn_kernel<<<dim3(BB * HH, SS / 64),    128, 0, stream>>>(qd, kd, vT, oc);
  proj_kernel<<<dim3(BB * SS / 64),        128, 0, stream>>>(oc, woT, out);
}
